// HumanObjectRelationModule_65755949301977
// MI455X (gfx1250) — compile-verified, matrix operand order checked by reading, not run on hardware
//
#include <hip/hip_runtime.h>
#include <hip/hip_bf16.h>
#include <math.h>

// Problem constants (from reference): M=512 gt boxes, N=1024 ctx boxes,
// G=16 groups, FDIM=1024 features, DIM_K=64 per group.
#define GNUM 16
#define MDIM 512
#define NDIM 1024
#define FDIM 1024

typedef __attribute__((ext_vector_type(16))) __bf16        v16bf;
typedef __attribute__((ext_vector_type(8)))  float         v8f;
typedef __attribute__((ext_vector_type(4)))  unsigned int  u32x4;
typedef int i32x4 __attribute__((vector_size(16)));   // matches async-LDS builtin param

union FragBF { v16bf v; u32x4 q[2]; };

// Use the gfx1250 async global->LDS copy path (ASYNCcnt-tracked) when the
// toolchain exposes it; otherwise fall back to load + ds_store.
#if __has_builtin(__builtin_amdgcn_global_load_async_to_lds_b128) && \
    __has_builtin(__builtin_amdgcn_s_wait_asynccnt)
#define USE_ASYNC_LDS 1
#else
#define USE_ASYNC_LDS 0
#endif

__device__ __forceinline__ unsigned short f2bf(float f) {
  unsigned int u = __float_as_uint(f);
  u += 0x7fffu + ((u >> 16) & 1u);     // round-to-nearest-even
  return (unsigned short)(u >> 16);
}

// ---------------------------------------------------------------------------
// f32 -> bf16 elementwise convert
// ---------------------------------------------------------------------------
__global__ void cvt_bf16_kernel(const float* __restrict__ in,
                                unsigned short* __restrict__ out, int n) {
  int i = blockIdx.x * blockDim.x + threadIdx.x;
  if (i < n) out[i] = f2bf(in[i]);
}

// ---------------------------------------------------------------------------
// Generic batched bf16 GEMM: C[b] = scale * A[b](MxK) @ B[b](NxK)^T + bias
// A, B bf16 row-major. Output f32 (Cf) or bf16 (Cbf).
//
// PRECONDITION (all launches below satisfy this): M % 128 == 0, N % 16 == 0,
// K % 32 == 0 — every wave is fully active, no exec masking around WMMA.
//
// Block = 256 threads (8 waves, wave32). Block tile 128(M) x 64(N); each wave
// owns a 16(M) x 64(N) strip. The 64x32 bf16 B panel for each k-step is shared
// by all 8 waves, so it is staged into LDS once per block (one b128 per
// thread) via the async global->LDS path, double-buffered so staging k+1
// overlaps the 4 WMMAs of step k. LDS rows padded to 40 ushorts (80B).
// Fragment layout per CDNA5 ISA 7.12.2 (16-bit A/B 16x32): lane L -> row L&15,
// K-half kh=L>>4: elements [0..7] = K kh*8.., [8..15] = K 16+kh*8.. .
// ---------------------------------------------------------------------------
__global__ void gemm_bf16_kernel(
    const unsigned short* __restrict__ A, const unsigned short* __restrict__ B,
    float* __restrict__ Cf, unsigned short* __restrict__ Cbf,
    const float* __restrict__ bias,
    int M, int N, int K, int lda, int ldb, int ldc,
    long long strideA, long long strideB, long long strideC, int strideBias,
    float scale)
{
  const int b = blockIdx.z;
  const unsigned short* Ab = A + (long long)b * strideA;
  const unsigned short* Bb = B + (long long)b * strideB;

  const int lane = threadIdx.x & 31;
  const int wave = threadIdx.x >> 5;                 // 0..7
  const int m0 = blockIdx.y * 128 + wave * 16;       // wave-uniform, always < M
  const int n0 = blockIdx.x * 64;

  const int r  = lane & 15;
  const int kh = lane >> 4;

  __shared__ __attribute__((aligned(16))) unsigned short lbs[2][64 * 40];

  // Staging: 256 threads x 16B = 64 rows x 64B (one 32-elem k-chunk per row).
  const int srow   = threadIdx.x >> 2;               // 0..63
  const int schunk = (threadIdx.x & 3) * 8;          // ushort offset 0/8/16/24
  const unsigned short* bStage = Bb + (long long)(n0 + srow) * ldb + schunk;
  unsigned short* sd0 = &lbs[0][srow * 40 + schunk];
  unsigned short* sd1 = &lbs[1][srow * 40 + schunk];

#if USE_ASYNC_LDS
#define STAGE_B128(dst, src)                                                   \
  __builtin_amdgcn_global_load_async_to_lds_b128(                              \
      (__attribute__((address_space(1))) i32x4*)(src),                         \
      (__attribute__((address_space(3))) i32x4*)(dst), 0, 0)
#else
#define STAGE_B128(dst, src) (*(u32x4*)(dst) = *(const u32x4*)(src))
#endif

  // Prologue: stage k=0 into buffer 0.
  STAGE_B128(sd0, bStage);
#if USE_ASYNC_LDS
  __builtin_amdgcn_s_wait_asynccnt(0);
#endif
  __syncthreads();

  v8f acc[4] = {};
  const unsigned short* aRow = Ab + (long long)(m0 + r) * lda;

  for (int k = 0; k < K; k += 32) {
    const int p = (k >> 5) & 1;
    if (k + 32 < K) {                                // stage next chunk
      STAGE_B128(p ? sd0 : sd1, bStage + k + 32);
    }
    FragBF af;
    af.q[0] = *(const u32x4*)(aRow + k + kh * 8);
    af.q[1] = *(const u32x4*)(aRow + k + kh * 8 + 16);
    const unsigned short* sb = p ? &lbs[1][0] : &lbs[0][0];
#pragma unroll
    for (int t = 0; t < 4; ++t) {
      const unsigned short* brow = sb + (t * 16 + r) * 40;
      FragBF bf;
      bf.q[0] = *(const u32x4*)(brow + kh * 8);
      bf.q[1] = *(const u32x4*)(brow + kh * 8 + 16);
      acc[t] = __builtin_amdgcn_wmma_f32_16x16x32_bf16(
          /*neg_a=*/false, af.v, /*neg_b=*/false, bf.v,
          /*c_mod=*/(short)0, acc[t], /*reuse_a=*/false, /*reuse_b=*/false);
    }
#if USE_ASYNC_LDS
    __builtin_amdgcn_s_wait_asynccnt(0);
#endif
    __syncthreads();
  }

  // C layout (ISA 7.12.2): VGPR rr of lane L -> M = rr + 8*(L>>4), N = L&15.
  const int cn = lane & 15;
  const int cm = (lane >> 4) * 8;
#pragma unroll
  for (int t = 0; t < 4; ++t) {
    int ncol = n0 + t * 16 + cn;
    if (ncol >= N) continue;
    float bv = bias ? bias[(long long)b * strideBias + ncol] : 0.0f;
#pragma unroll
    for (int rr = 0; rr < 8; ++rr) {
      int row = m0 + cm + rr;
      float val = acc[t][rr] * scale + bv;
      long long ci = (long long)b * strideC + (long long)row * ldc + ncol;
      if (Cf) Cf[ci] = val;
      else    Cbf[ci] = f2bf(val);
    }
  }
#undef STAGE_B128
}

// ---------------------------------------------------------------------------
// Position-bias kernel: for each (row, col) box pair compute the 64-dim
// sin/cos embedding ONCE, relu it, and for all 16 groups write
//   Bout[g,row,col] = S(g,row,col) + log(max(relu(emb) . w_pos[g] + b, 1e-6))
// transS=0: S read as (G,Mrow,Ncol). transS=1: S read as (G,Ncol,Mrow).
// ---------------------------------------------------------------------------
__global__ void pos_bias_kernel(
    const float* __restrict__ S, float* __restrict__ Bout,
    const float* __restrict__ rbox, const float* __restrict__ cbox,
    const float* __restrict__ wpos, const float* __restrict__ bpos,
    int Mrow, int Ncol, int transS)
{
  __shared__ float wsm[GNUM * 64];
  __shared__ float bsm[GNUM];
  for (int i = threadIdx.x; i < GNUM * 64; i += blockDim.x) wsm[i] = wpos[i];
  if (threadIdx.x < GNUM) bsm[threadIdx.x] = bpos[threadIdx.x];
  __syncthreads();

  long long idx = (long long)blockIdx.x * blockDim.x + threadIdx.x;
  long long total = (long long)Mrow * Ncol;
  if (idx >= total) return;
  int row = (int)(idx / Ncol);
  int col = (int)(idx % Ncol);

  float x1 = rbox[row * 4 + 0], y1 = rbox[row * 4 + 1];
  float x2 = rbox[row * 4 + 2], y2 = rbox[row * 4 + 3];
  float bw = x2 - x1 + 1.0f, bh = y2 - y1 + 1.0f;
  float cx = 0.5f * (x1 + x2), cy = 0.5f * (y1 + y2);

  float u1 = cbox[col * 4 + 0], v1 = cbox[col * 4 + 1];
  float u2 = cbox[col * 4 + 2], v2 = cbox[col * 4 + 3];
  float cbw = u2 - u1 + 1.0f, cbh = v2 - v1 + 1.0f;
  float ccx = 0.5f * (u1 + u2), ccy = 0.5f * (v1 + v2);

  float pos4[4];
  pos4[0] = __logf(fmaxf(fabsf((cx - ccx) / bw), 1e-3f));
  pos4[1] = __logf(fmaxf(fabsf((cy - ccy) / bh), 1e-3f));
  pos4[2] = __logf(cbw / bw);
  pos4[3] = __logf(cbh / bh);

  // emb[f*16 + j] = sin(div), emb[f*16 + 8 + j] = cos(div); relu applied.
  float emb[64];
#pragma unroll
  for (int f = 0; f < 4; ++f) {
#pragma unroll
    for (int j = 0; j < 8; ++j) {
      float invd = __powf(1000.0f, -0.125f * (float)j);
      float arg = 100.0f * pos4[f] * invd;
      float s, c;
      __sincosf(arg, &s, &c);
      emb[f * 16 + j]     = fmaxf(s, 0.0f);
      emb[f * 16 + 8 + j] = fmaxf(c, 0.0f);
    }
  }

  long long plane = (long long)Mrow * Ncol;
#pragma unroll 1
  for (int g = 0; g < GNUM; ++g) {
    float p = bsm[g];
#pragma unroll
    for (int d = 0; d < 64; ++d) p += emb[d] * wsm[g * 64 + d];
    float lb = __logf(fmaxf(p, 1e-6f));
    long long si = transS
        ? ((long long)g * plane + (long long)col * Mrow + row)
        : ((long long)g * plane + (long long)row * Ncol + col);
    Bout[(long long)g * plane + (long long)row * Ncol + col] = S[si] + lb;
  }
}

// ---------------------------------------------------------------------------
// Row softmax (f32 in) -> bf16 probabilities. One 256-thread block per row.
// cols <= 1024.
// ---------------------------------------------------------------------------
__global__ void softmax_bf16_kernel(const float* __restrict__ Bin,
                                    unsigned short* __restrict__ P, int cols)
{
  long long row = blockIdx.x;
  const float* x = Bin + row * cols;
  unsigned short* y = P + row * cols;
  __shared__ float red[256];

  float lmax = -INFINITY;
  for (int i = threadIdx.x; i < cols; i += 256) lmax = fmaxf(lmax, x[i]);
  red[threadIdx.x] = lmax; __syncthreads();
  for (int s = 128; s > 0; s >>= 1) {
    if (threadIdx.x < s) red[threadIdx.x] = fmaxf(red[threadIdx.x], red[threadIdx.x + s]);
    __syncthreads();
  }
  float m = red[0]; __syncthreads();

  float vals[4];
  int cnt = 0;
  float lsum = 0.0f;
  for (int i = threadIdx.x; i < cols; i += 256) {
    float e = __expf(x[i] - m);
    vals[cnt++] = e;
    lsum += e;
  }
  red[threadIdx.x] = lsum; __syncthreads();
  for (int s = 128; s > 0; s >>= 1) {
    if (threadIdx.x < s) red[threadIdx.x] += red[threadIdx.x + s];
    __syncthreads();
  }
  float inv = 1.0f / red[0];

  cnt = 0;
  for (int i = threadIdx.x; i < cols; i += 256) y[i] = f2bf(vals[cnt++] * inv);
}

// ---------------------------------------------------------------------------
// Host-side orchestration
// ---------------------------------------------------------------------------
extern "C" void kernel_launch(void* const* d_in, const int* in_sizes, int n_in,
                              void* d_out, int out_size, void* d_ws, size_t ws_size,
                              hipStream_t stream) {
  const float* feat      = (const float*)d_in[0];   // (512,1024)
  const float* ctx_feat  = (const float*)d_in[1];   // (1024,1024)
  const float* box       = (const float*)d_in[2];   // (512,4)
  const float* ctx_box   = (const float*)d_in[3];   // (1024,4)
  const float* w_fc_gt   = (const float*)d_in[4];   // (1024,1024)
  const float* b_fc_gt   = (const float*)d_in[5];
  const float* w_fc_ctx  = (const float*)d_in[6];
  const float* b_fc_ctx  = (const float*)d_in[7];
  const float* w_pos_gt  = (const float*)d_in[8];   // (16,64)
  const float* b_pos_gt  = (const float*)d_in[9];
  const float* w_pos_ctx = (const float*)d_in[10];
  const float* b_pos_ctx = (const float*)d_in[11];
  const float* w_conv    = (const float*)d_in[12];  // (16,64,1024) -> flat (1024,1024)
  const float* b_conv    = (const float*)d_in[13];

  float* out_gt  = (float*)d_out;                       // (512,1024)
  float* out_ctx = (float*)d_out + (size_t)MDIM * FDIM; // (1024,1024)

  char* ws = (char*)d_ws;
  size_t off = 0;
  auto alloc = [&](size_t bytes) -> void* {
    void* p = ws + off;
    off = (off + bytes + 255) & ~(size_t)255;
    return p;
  };

  unsigned short* featbf = (unsigned short*)alloc((size_t)MDIM * FDIM * 2);
  unsigned short* ctxbf  = (unsigned short*)alloc((size_t)NDIM * FDIM * 2);
  unsigned short* wqbf   = (unsigned short*)alloc((size_t)FDIM * FDIM * 2);
  unsigned short* wkbf   = (unsigned short*)alloc((size_t)FDIM * FDIM * 2);
  unsigned short* wcbf   = (unsigned short*)alloc((size_t)FDIM * FDIM * 2);
  unsigned short* Qbf    = (unsigned short*)alloc((size_t)MDIM * FDIM * 2);
  unsigned short* Kbf    = (unsigned short*)alloc((size_t)NDIM * FDIM * 2);
  unsigned short* VTbf   = (unsigned short*)alloc((size_t)FDIM * NDIM * 2); // (1024, N)
  unsigned short* UTbf   = (unsigned short*)alloc((size_t)FDIM * MDIM * 2); // (1024, M)
  float* S  = (float*)alloc((size_t)GNUM * MDIM * NDIM * 4);  // (G,M,N)
  float* B1 = (float*)alloc((size_t)GNUM * MDIM * NDIM * 4);  // (G,M,N)
  float* B2 = (float*)alloc((size_t)GNUM * NDIM * MDIM * 4);  // (G,N,M)
  unsigned short* P1 = (unsigned short*)alloc((size_t)GNUM * MDIM * NDIM * 2);
  unsigned short* P2 = (unsigned short*)alloc((size_t)GNUM * NDIM * MDIM * 2);

  // 1) convert activations + weights to bf16
  {
    int n;
    n = MDIM * FDIM; cvt_bf16_kernel<<<(n + 255) / 256, 256, 0, stream>>>(feat, featbf, n);
    n = NDIM * FDIM; cvt_bf16_kernel<<<(n + 255) / 256, 256, 0, stream>>>(ctx_feat, ctxbf, n);
    n = FDIM * FDIM; cvt_bf16_kernel<<<(n + 255) / 256, 256, 0, stream>>>(w_fc_gt, wqbf, n);
    n = FDIM * FDIM; cvt_bf16_kernel<<<(n + 255) / 256, 256, 0, stream>>>(w_fc_ctx, wkbf, n);
    n = FDIM * FDIM; cvt_bf16_kernel<<<(n + 255) / 256, 256, 0, stream>>>(w_conv, wcbf, n);
  }

  dim3 blk(256);
  // 2) projections (bf16 out, f32 WMMA accumulate)
  // Q = feat @ w_fc_gt^T + b          -> (M,1024) bf16
  gemm_bf16_kernel<<<dim3(FDIM / 64, MDIM / 128, 1), blk, 0, stream>>>(
      featbf, wqbf, nullptr, Qbf, b_fc_gt,
      MDIM, FDIM, FDIM, FDIM, FDIM, FDIM, 0, 0, 0, 0, 1.0f);
  // K = ctx_feat @ w_fc_ctx^T + b     -> (N,1024) bf16
  gemm_bf16_kernel<<<dim3(FDIM / 64, NDIM / 128, 1), blk, 0, stream>>>(
      ctxbf, wkbf, nullptr, Kbf, b_fc_ctx,
      NDIM, FDIM, FDIM, FDIM, FDIM, FDIM, 0, 0, 0, 0, 1.0f);
  // VT = w_conv_flat @ ctx_feat^T     -> (1024, N) bf16 (fused grouped conv)
  gemm_bf16_kernel<<<dim3(NDIM / 64, FDIM / 128, 1), blk, 0, stream>>>(
      wcbf, ctxbf, nullptr, VTbf, nullptr,
      FDIM, NDIM, FDIM, FDIM, FDIM, NDIM, 0, 0, 0, 0, 1.0f);
  // UT = w_conv_flat @ feat^T         -> (1024, M) bf16
  gemm_bf16_kernel<<<dim3(MDIM / 64, FDIM / 128, 1), blk, 0, stream>>>(
      wcbf, featbf, nullptr, UTbf, nullptr,
      FDIM, MDIM, FDIM, FDIM, FDIM, MDIM, 0, 0, 0, 0, 1.0f);

  // 3) scores: S[g] = (Q_g @ K_g^T) / sqrt(64)   (batched over g, K=64)
  gemm_bf16_kernel<<<dim3(NDIM / 64, MDIM / 128, GNUM), blk, 0, stream>>>(
      Qbf, Kbf, S, nullptr, nullptr,
      MDIM, NDIM, 64, FDIM, FDIM, NDIM,
      64, 64, (long long)MDIM * NDIM, 0, 0.125f);

  // 4) add log positional bias (both directions; embedding computed once per pair)
  {
    long long tot1 = (long long)MDIM * NDIM;
    pos_bias_kernel<<<(unsigned)((tot1 + 255) / 256), 256, 0, stream>>>(
        S, B1, box, ctx_box, w_pos_gt, b_pos_gt, MDIM, NDIM, 0);
    long long tot2 = (long long)NDIM * MDIM;
    pos_bias_kernel<<<(unsigned)((tot2 + 255) / 256), 256, 0, stream>>>(
        S, B2, ctx_box, box, w_pos_ctx, b_pos_ctx, NDIM, MDIM, 1);
  }

  // 5) softmax -> bf16 probabilities
  softmax_bf16_kernel<<<GNUM * MDIM, 256, 0, stream>>>(B1, P1, NDIM);
  softmax_bf16_kernel<<<GNUM * NDIM, 256, 0, stream>>>(B2, P2, MDIM);

  // 6) outputs: out_gt[m, g*64+o] = P1[g,m,:] . VT[g*64+o,:] + b_conv
  gemm_bf16_kernel<<<dim3(1, MDIM / 128, GNUM), blk, 0, stream>>>(
      P1, VTbf, out_gt, nullptr, b_conv,
      MDIM, 64, NDIM, NDIM, NDIM, FDIM,
      (long long)MDIM * NDIM, (long long)64 * NDIM, 64, 64, 1.0f);
  //          out_ctx[n, g*64+o] = P2[g,n,:] . UT[g*64+o,:] + b_conv
  gemm_bf16_kernel<<<dim3(1, NDIM / 128, GNUM), blk, 0, stream>>>(
      P2, UTbf, out_ctx, nullptr, b_conv,
      NDIM, 64, MDIM, MDIM, MDIM, FDIM,
      (long long)NDIM * MDIM, (long long)64 * MDIM, 64, 64, 1.0f);
}
